// LearnableFeaturePermute_38431367364888
// MI455X (gfx1250) — compile-verified, matrix operand order checked by reading, not run on hardware
//
#include <hip/hip_runtime.h>
#include <hip/hip_bf16.h>
#include <math.h>

typedef __attribute__((ext_vector_type(4)))  _Float16 v4h;
typedef __attribute__((ext_vector_type(8)))  _Float16 v8h;
typedef __attribute__((ext_vector_type(16))) _Float16 v16h;
typedef __attribute__((ext_vector_type(8)))  float    v8f;

#define F 1024
#define BROWS 65536
#define SINK_BLOCKS 64
#define NUM_ITERS 10

// workspace layout
#define WS_LOGW_OFF  1024u
#define WS_PT_OFF    (1024u + (unsigned)(F) * (unsigned)(F) * 4u)

// GEMM tiling
#define GM   128      // M rows per block (A panel resident in LDS)
#define NCH  128      // N chunk
#define KC   32       // K step (WMMA K)
#define ASTR 1032     // A LDS row stride in halves (1024 + 8 pad -> bank-conflict-free frags)
#define BSTR 40       // B LDS row stride in halves (32 + 8 pad)
#define A_LDS_HALVES (GM * ASTR)
#define B_LDS_HALVES (NCH * BSTR)

// CDNA5 async global->LDS path (ASYNCcnt-tracked, no VGPR round trip)
#if defined(__has_builtin)
#if __has_builtin(__builtin_amdgcn_global_load_async_to_lds_b128) && \
    __has_builtin(__builtin_amdgcn_s_wait_asynccnt)
#define LFP_ASYNC_LDS 1
#endif
#endif

#ifdef LFP_ASYNC_LDS
// builtin signature (from hipcc diagnostic): param0 = int4* in global (__device__/AS1)
typedef int v4i_vs __attribute__((vector_size(4 * sizeof(int))));
typedef __attribute__((address_space(1))) v4i_vs gas_v4i;   // global int4
typedef __attribute__((address_space(3))) v4i_vs lds_v4i;   // LDS int4
// copy 32 bytes (one B-tile row-half pair) global -> LDS asynchronously
static __device__ inline void lfp_async_cp32(const _Float16* g, _Float16* l) {
  __builtin_amdgcn_global_load_async_to_lds_b128((gas_v4i*)g, (lds_v4i*)l, 0, 0);
  __builtin_amdgcn_global_load_async_to_lds_b128((gas_v4i*)g, (lds_v4i*)l, 16, 0);
}
#endif

static __device__ inline v16h cat16(v8h lo, v8h hi) {
  return __builtin_shufflevector(lo, hi, 0,1,2,3,4,5,6,7,8,9,10,11,12,13,14,15);
}

// ---------------- barrier init (ws is poisoned once; re-zero every call) ----
__global__ void lfp_init_bar(int* bar) {
  if (threadIdx.x < 64) bar[threadIdx.x] = 0;
}

// ---------------- grid-wide barrier (persistent kernel, 64 resident blocks) -
__device__ inline void grid_barrier(int* bar, int idx, int nblk) {
  __threadfence();
  __syncthreads();
  if (threadIdx.x == 0) {
    atomicAdd(&bar[idx], 1);
    while (atomicAdd(&bar[idx], 0) < nblk) __builtin_amdgcn_s_sleep(1);
  }
  __syncthreads();
  __threadfence();
}

__device__ inline float block_red_max(float v, float* scratch, int tid) {
  #pragma unroll
  for (int o = 16; o > 0; o >>= 1) v = fmaxf(v, __shfl_xor(v, o, 32));
  if ((tid & 31) == 0) scratch[tid >> 5] = v;
  __syncthreads();
  float r = scratch[0];
  #pragma unroll
  for (int i = 1; i < 8; ++i) r = fmaxf(r, scratch[i]);
  __syncthreads();
  return r;
}

__device__ inline float block_red_sum(float v, float* scratch, int tid) {
  #pragma unroll
  for (int o = 16; o > 0; o >>= 1) v += __shfl_xor(v, o, 32);
  if ((tid & 31) == 0) scratch[tid >> 5] = v;
  __syncthreads();
  float r = scratch[0];
  #pragma unroll
  for (int i = 1; i < 8; ++i) r += scratch[i];
  __syncthreads();
  return r;
}

// ---------------- fused Sinkhorn (10 iters) + exp + transpose to f16 --------
__global__ __launch_bounds__(256)
void lfp_sinkhorn(const float* __restrict__ logits, float* __restrict__ logW,
                  _Float16* __restrict__ Pt, int* bar) {
  __shared__ float sA[256];
  __shared__ float sB[256];
  __shared__ _Float16 tile[128 * 130];

  const int tid = threadIdx.x;
  const int bid = blockIdx.x;

  // copy logits -> working buffer (TEMPERATURE == 1.0)
  const int total4 = F * F / 4;
  for (int i = bid * 256 + tid; i < total4; i += SINK_BLOCKS * 256)
    ((float4*)logW)[i] = ((const float4*)logits)[i];
  grid_barrier(bar, 0, SINK_BLOCKS);

  for (int it = 0; it < NUM_ITERS; ++it) {
    // ---- row log-softmax: one block-row at a time, values live in registers
    for (int j = 0; j < 16; ++j) {
      const int r = bid * 16 + j;
      float4 v = ((float4*)logW)[r * (F / 4) + tid];
      float mx = fmaxf(fmaxf(v.x, v.y), fmaxf(v.z, v.w));
      mx = block_red_max(mx, sA, tid);
      float s = __expf(v.x - mx) + __expf(v.y - mx) + __expf(v.z - mx) + __expf(v.w - mx);
      s = block_red_sum(s, sA, tid);
      const float nrm = mx + __logf(s);
      v.x -= nrm; v.y -= nrm; v.z -= nrm; v.w -= nrm;
      ((float4*)logW)[r * (F / 4) + tid] = v;
    }
    grid_barrier(bar, 1 + 2 * it, SINK_BLOCKS);

    // ---- column log-softmax: block owns 16 columns, 16 threads per column
    {
      const int tx = tid & 15, ty = tid >> 4;
      const int c = bid * 16 + tx;
      float m = -INFINITY, s = 0.f;
      for (int r = ty; r < F; r += 16) {
        const float vv = logW[r * F + c];
        const float nm = fmaxf(m, vv);
        s = s * __expf(m - nm) + __expf(vv - nm);
        m = nm;
      }
      sA[tid] = m; sB[tid] = s;
      __syncthreads();
      float M = sA[tx], S = sB[tx];
      #pragma unroll
      for (int i = 1; i < 16; ++i) {
        const float m2 = sA[i * 16 + tx], s2 = sB[i * 16 + tx];
        const float nm = fmaxf(M, m2);
        S = S * __expf(M - nm) + s2 * __expf(m2 - nm);
        M = nm;
      }
      const float nrm = M + __logf(S);
      __syncthreads();
      for (int r = ty; r < F; r += 16) logW[r * F + c] -= nrm;
    }
    grid_barrier(bar, 2 + 2 * it, SINK_BLOCKS);
  }

  // ---- P = exp(logW), write transposed f16: Pt[n][k] = exp(logW[k][n])
  {
    const int tr = bid >> 3, tc = bid & 7;   // 8x8 grid of 128x128 tiles
    const int lr = tid >> 7;                 // 0..1
    const int lc = tid & 127;                // 0..127
    for (int sweep = 0; sweep < 64; ++sweep) {
      const int rr = sweep * 2 + lr;
      const float v = logW[(tr * 128 + rr) * F + tc * 128 + lc];
      tile[lc * 130 + rr] = (_Float16)__expf(v);
    }
    __syncthreads();
    for (int sweep = 0; sweep < 64; ++sweep) {
      const int nn = sweep * 2 + lr;
      Pt[(size_t)(tc * 128 + nn) * F + tr * 128 + lc] = tile[nn * 130 + lc];
    }
  }
}

// ---------------- WMMA GEMM: out = x @ P, A panel resident in 320KB LDS -----
__global__ __launch_bounds__(256)
void lfp_gemm(const float* __restrict__ x, const _Float16* __restrict__ Pt,
              float* __restrict__ out) {
  extern __shared__ _Float16 smem[];
  _Float16* As  = smem;                       // [128][1024] f16, stride 1032
  _Float16* Bs0 = smem + A_LDS_HALVES;        // [128 n][32 k] f16, stride 40
  _Float16* Bs1 = Bs0 + B_LDS_HALVES;

  const int tid = threadIdx.x;
  const int bM  = blockIdx.x * GM;

  // ---- stage full A panel: 128 rows x 1024 K of x, fp32 -> f16, one HBM pass
  {
    const float4* xv = (const float4*)x;
    for (int m = 0; m < GM; ++m) {
      const float4 v = xv[(size_t)(bM + m) * (F / 4) + tid];
      v4h h; h[0] = (_Float16)v.x; h[1] = (_Float16)v.y;
             h[2] = (_Float16)v.z; h[3] = (_Float16)v.w;
      *(v4h*)(As + m * ASTR + tid * 4) = h;
    }
  }
  __syncthreads();

  const int lane  = tid & 31;
  const int wv    = tid >> 5;
  const int wm    = wv & 3;        // 4 waves across M (32 rows each)
  const int wn    = wv >> 2;       // 2 waves across N (64 cols each)
  const int laneM = lane & 15;
  const int hi    = lane >> 4;     // 0/1 -> K-half select per WMMA layout

  const int prow = tid >> 1;       // B staging: 0..127 (n row)
  const int psel = tid & 1;        // which 16-half chunk of the 32-K row

  for (int nc = 0; nc < F / NCH; ++nc) {
    const int ncb = nc * NCH;
    v8f acc[2][4];
    #pragma unroll
    for (int sm = 0; sm < 2; ++sm)
      #pragma unroll
      for (int sn = 0; sn < 4; ++sn) {
        v8f z = {0.f, 0.f, 0.f, 0.f, 0.f, 0.f, 0.f, 0.f};
        acc[sm][sn] = z;
      }

    // stage first B tile (Pt is hot in L2: only 2 MB)
    {
      const _Float16* src = Pt + (size_t)(ncb + prow) * F + psel * 16;
      _Float16* dst = Bs0 + prow * BSTR + psel * 16;
#ifdef LFP_ASYNC_LDS
      lfp_async_cp32(src, dst);
      __builtin_amdgcn_s_wait_asynccnt(0);
#else
      const v8h g0 = *(const v8h*)src;
      const v8h g1 = *(const v8h*)(src + 8);
      *(v8h*)dst = g0; *(v8h*)(dst + 8) = g1;
#endif
    }
    __syncthreads();

    #pragma unroll 1
    for (int ks = 0; ks < F / KC; ++ks) {
      _Float16* Bcur = (ks & 1) ? Bs1 : Bs0;
      _Float16* Bnxt = (ks & 1) ? Bs0 : Bs1;
      const bool pf = (ks + 1 < F / KC);

#ifdef LFP_ASYNC_LDS
      // issue async DMA for the next tile up front; it lands during the WMMAs
      if (pf) {
        const _Float16* src = Pt + (size_t)(ncb + prow) * F + (ks + 1) * KC + psel * 16;
        lfp_async_cp32(src, Bnxt + prow * BSTR + psel * 16);
      }
#else
      v8h g0, g1;
      if (pf) {  // prefetch next B tile into registers
        const _Float16* src = Pt + (size_t)(ncb + prow) * F + (ks + 1) * KC + psel * 16;
        g0 = *(const v8h*)src;
        g1 = *(const v8h*)(src + 8);
      }
#endif

      // load fragments from LDS (two ds_load_b128 each, bank-conflict-free)
      v16h afrag[2], bfrag[4];
      #pragma unroll
      for (int sm = 0; sm < 2; ++sm) {
        const _Float16* ap = As + (wm * 32 + sm * 16 + laneM) * ASTR + ks * KC + hi * 8;
        afrag[sm] = cat16(*(const v8h*)ap, *(const v8h*)(ap + 16));
      }
      #pragma unroll
      for (int sn = 0; sn < 4; ++sn) {
        const _Float16* bp = Bcur + (wn * 64 + sn * 16 + laneM) * BSTR + hi * 8;
        bfrag[sn] = cat16(*(const v8h*)bp, *(const v8h*)(bp + 16));
      }

      #pragma unroll
      for (int sm = 0; sm < 2; ++sm)
        #pragma unroll
        for (int sn = 0; sn < 4; ++sn)
          acc[sm][sn] = __builtin_amdgcn_wmma_f32_16x16x32_f16(
              false, afrag[sm], false, bfrag[sn], (short)0, acc[sm][sn], false, false);

#ifdef LFP_ASYNC_LDS
      if (pf) __builtin_amdgcn_s_wait_asynccnt(0);
#else
      if (pf) {  // commit prefetched tile to the other buffer
        _Float16* dst = Bnxt + prow * BSTR + psel * 16;
        *(v8h*)dst = g0; *(v8h*)(dst + 8) = g1;
      }
#endif
      __syncthreads();
    }

    // ---- store C: VGPR r holds (M = r + 8*hi, N = laneM) per ISA C layout
    #pragma unroll
    for (int sm = 0; sm < 2; ++sm) {
      const int row0 = bM + wm * 32 + sm * 16 + hi * 8;
      #pragma unroll
      for (int sn = 0; sn < 4; ++sn) {
        const int col = ncb + wn * 64 + sn * 16 + laneM;
        #pragma unroll
        for (int r = 0; r < 8; ++r)
          out[(size_t)(row0 + r) * F + col] = acc[sm][sn][r];
      }
    }
  }
}

extern "C" void kernel_launch(void* const* d_in, const int* in_sizes, int n_in,
                              void* d_out, int out_size, void* d_ws, size_t ws_size,
                              hipStream_t stream) {
  (void)in_sizes; (void)n_in; (void)out_size; (void)ws_size;
  const float* x      = (const float*)d_in[0];   // [65536,1024] fp32
  const float* logits = (const float*)d_in[1];   // [1024,1024]  fp32
  float* out = (float*)d_out;

  int*      bar  = (int*)d_ws;                                  // 64 ints
  float*    logW = (float*)((char*)d_ws + WS_LOGW_OFF);         // 4 MB
  _Float16* Pt   = (_Float16*)((char*)d_ws + WS_PT_OFF);        // 2 MB (P^T f16)

  lfp_init_bar<<<1, 64, 0, stream>>>(bar);
  lfp_sinkhorn<<<SINK_BLOCKS, 256, 0, stream>>>(logits, logW, Pt, bar);

  const size_t lds_bytes = (size_t)(A_LDS_HALVES + 2 * B_LDS_HALVES) * sizeof(_Float16); // ~278 KB
  lfp_gemm<<<BROWS / GM, 256, lds_bytes, stream>>>(x, Pt, out);
}
